// BUY_Net_29635274342639
// MI455X (gfx1250) — matrix operand order repaired and checked by reading, not compiled, not run on hardware
//
#include <hip/hip_runtime.h>
#include <hip/hip_bf16.h>
#include <math.h>

typedef __bf16 bf16;
typedef __attribute__((ext_vector_type(16))) __bf16 v16bf;
typedef __attribute__((ext_vector_type(8)))  float  v8f;

typedef unsigned int u32x4 __attribute__((ext_vector_type(4)));
typedef int          i32x4 __attribute__((ext_vector_type(4)));
typedef int          i32x8 __attribute__((ext_vector_type(8)));

#if defined(__has_builtin)
#if __has_builtin(__builtin_amdgcn_tensor_load_to_lds) && __has_builtin(__builtin_amdgcn_s_wait_tensorcnt)
#define HAVE_TDM 1
#endif
#endif
#ifndef HAVE_TDM
#define HAVE_TDM 0
#endif

namespace {
constexpr int cG = 2048;   // graphs
constexpr int cN = 64;     // nodes per graph
constexpr int cE = 256;    // edges per graph
constexpr int cD = 128;    // embed dim
constexpr int KP1 = 58, KP2 = 53, KP3 = 48;
}

// ---------- bf16 helpers (RNE convert, bit-exact, no libcall) ----------
__device__ __forceinline__ bf16 f2bf(float f) {
  union { float f; unsigned u; } a; a.f = f;
  unsigned r = a.u + 0x7FFFu + ((a.u >> 16) & 1u);
  union { unsigned short s; bf16 b; } o; o.s = (unsigned short)(r >> 16);
  return o.b;
}
__device__ __forceinline__ float bf2f(bf16 b) {
  union { unsigned short s; bf16 b; } i; i.b = b;
  union { unsigned u; float f; } o; o.u = ((unsigned)i.s) << 16;
  return o.f;
}

// ---------- WMMA fragment loaders (ISA 7.12.2 layouts, wave32) ----------
// A: 16x32 bf16 tile from row-major matrix X[ld], rows mbase..mbase+15, cols k0..k0+31.
// lanes 0-15: M=lane, K in {0..7,16..23}; lanes 16-31: same M, K in {8..15,24..31}.
__device__ __forceinline__ v16bf load_a_frag(const bf16* X, int ld, int mbase, int k0, int lane) {
  int m = mbase + (lane & 15);
  int g = lane >> 4;
  v16bf a;
#pragma unroll
  for (int e = 0; e < 16; ++e) {
    int v = e >> 1, s = e & 1;
    int k = (v < 4) ? (v * 2 + s + g * 8) : (16 + (v - 4) * 2 + s + g * 8);
    a[e] = X[(size_t)m * ld + k0 + k];
  }
  return a;
}
// B: 32x16 bf16 tile from row-major weight W[ld] (K-major), rows k0..k0+31, cols n0..n0+15.
__device__ __forceinline__ v16bf load_b_frag(const bf16* W, int ld, int k0, int n0, int lane) {
  int n = n0 + (lane & 15);
  int g = lane >> 4;
  v16bf b;
#pragma unroll
  for (int e = 0; e < 16; ++e)
    b[e] = W[(size_t)(k0 + (e < 8 ? g * 8 + e : 16 + g * 8 + (e - 8))) * ld + n];
  return b;
}
__device__ __forceinline__ v8f wmma_bf16(v16bf a, v16bf b, v8f c) {
  return __builtin_amdgcn_wmma_f32_16x16x32_bf16(false, a, false, b, (short)0, c, false, false);
}

// ---------- small utility kernels ----------
__global__ void k_cvt(const float* __restrict__ src, bf16* __restrict__ dst, int n) {
  int t = blockIdx.x * blockDim.x + threadIdx.x;
  if (t < n) dst[t] = f2bf(src[t]);
}
__global__ void k_zero(float* __restrict__ p, int n) {
  int t = blockIdx.x * blockDim.x + threadIdx.x;
  if (t < n) p[t] = 0.f;
}

// ======================================================================
// Layer 1: gather embeddings + GraphConv(256->128) + ReLU + TopK + readout
// One block (8 waves) per graph. LDS-resident graph.
// ======================================================================
__global__ __launch_bounds__(256)
void k_layer1(const int* __restrict__ item_id, const int* __restrict__ category,
              const int* __restrict__ edge_local,
              const float* __restrict__ emb_item, const float* __restrict__ emb_cat,
              const bf16* __restrict__ Wr, const float* __restrict__ bias,
              const bf16* __restrict__ Wo, const float* __restrict__ p,
              int* __restrict__ src_out, int* __restrict__ dst_out, int* __restrict__ msk_out,
              bf16* __restrict__ xout, float* __restrict__ h) {
  // LDS layout (aliased regions): sX 32K | sAggBf 32K | region 64K | small
  __shared__ __align__(16) unsigned char smem[132128];
  bf16*  sX     = (bf16*)smem;                       // [64][256]
  bf16*  sAggBf = (bf16*)(smem + 32768);             // [64][256]
  float* sAgg   = (float*)(smem + 65536);            // [64][256] f32 (pre-WMMA)
  float* sY     = (float*)(smem + 65536);            // [64][128] aliases sAgg (post)
  float* sXk    = (float*)(smem + 65536 + 32768);    // [64][128] aliases sAgg hi half
  float* sS     = (float*)(smem + 131072);           // [64]
  float* sVal   = (float*)(smem + 131072 + 256);     // [64]
  int*   sIdx   = (int*)  (smem + 131072 + 512);     // [64]
  int*   sInv   = (int*)  (smem + 131072 + 768);     // [64]
  float* sPn    = (float*)(smem + 131072 + 1024);

  const int g = blockIdx.x;
  const int tid = threadIdx.x;
  const int lane = tid & 31, wv = tid >> 5;

  // gather node features: [64][256] bf16
  for (int idx = tid; idx < 64 * 256; idx += 256) {
    int j = idx >> 8, c = idx & 255;
    float v;
    if (c < 128) v = emb_item[(size_t)item_id[g * cN + j] * cD + c];
    else         v = emb_cat [(size_t)category[g * cN + j] * cD + (c - 128)];
    sX[idx] = f2bf(v);
  }
  for (int i = tid; i < 64 * 256; i += 256) sAgg[i] = 0.f;
  __syncthreads();

  // edge scatter-add into LDS (ds_add_f32)
  for (int idx = tid; idx < cE * 256; idx += 256) {
    int e = idx >> 8, c = idx & 255;
    int s0 = edge_local[(size_t)g * cE + e];
    int d0 = edge_local[(size_t)cG * cE + (size_t)g * cE + e];
    atomicAdd(&sAgg[d0 * 256 + c], bf2f(sX[s0 * 256 + c]));
  }
  __syncthreads();
  for (int i = tid; i < 64 * 256; i += 256) sAggBf[i] = f2bf(sAgg[i]);
  __syncthreads();

  // y = relu(agg@Wr + x@Wo + b) via WMMA: 4 M-tiles x 8 N-tiles, K=256
  for (int t = wv; t < 32; t += 8) {
    int mt = t >> 3, nt = t & 7;
    v8f acc = {};
#pragma unroll
    for (int kk = 0; kk < 256; kk += 32) {
      acc = wmma_bf16(load_a_frag(sAggBf, 256, mt * 16, kk, lane),
                      load_b_frag(Wr, 128, kk, nt * 16, lane), acc);
      acc = wmma_bf16(load_a_frag(sX, 256, mt * 16, kk, lane),
                      load_b_frag(Wo, 128, kk, nt * 16, lane), acc);
    }
    int n = nt * 16 + (lane & 15);
    int mb = mt * 16 + (lane >> 4) * 8;
    float bv = bias[n];
#pragma unroll
    for (int r = 0; r < 8; ++r) {
      float v = acc[r] + bv;
      sY[(mb + r) * 128 + n] = v > 0.f ? v : 0.f;
    }
  }
  __syncthreads();

  // scores and top-k (n=64, k=58)
  if (tid == 0) {
    float s = 0.f;
    for (int c = 0; c < 128; ++c) s += p[c] * p[c];
    *sPn = sqrtf(s);
  }
  __syncthreads();
  if (tid < 64) {
    float s = 0.f;
    for (int c = 0; c < 128; ++c) s += sY[tid * 128 + c] * p[c];
    sS[tid] = s / *sPn;
  }
  __syncthreads();
  if (tid == 0) {
    for (int j = 0; j < 64; ++j) sInv[j] = -1;
    for (int i = 0; i < KP1; ++i) {
      float best = -1e30f; int bj = 0;
      for (int j = 0; j < 64; ++j) if (sS[j] > best) { best = sS[j]; bj = j; }
      sIdx[i] = bj; sVal[i] = best; sInv[bj] = i; sS[bj] = -1e30f;
    }
  }
  __syncthreads();

  // scale selected rows, pad to 64, emit bf16 for next layer
  for (int idx = tid; idx < 64 * 128; idx += 256) {
    int i = idx >> 7, c = idx & 127;
    float v = 0.f;
    if (i < KP1) v = sY[sIdx[i] * 128 + c] * tanhf(sVal[i]);
    sXk[idx] = v;
    xout[(size_t)g * 64 * 128 + idx] = f2bf(v);
  }
  __syncthreads();

  // readout (max | mean) accumulated into h[g][256]
  if (tid < 128) {
    int c = tid;
    float mx = -1e30f, sm = 0.f;
    for (int i = 0; i < KP1; ++i) { float v = sXk[i * 128 + c]; mx = fmaxf(mx, v); sm += v; }
    h[(size_t)g * 256 + c]       += mx;
    h[(size_t)g * 256 + 128 + c] += sm / (float)KP1;
  }

  // edge remap
  for (int e = tid; e < cE; e += 256) {
    int s0 = edge_local[(size_t)g * cE + e];
    int d0 = edge_local[(size_t)cG * cE + (size_t)g * cE + e];
    int ns = sInv[s0], nd = sInv[d0];
    int m = (ns >= 0 && nd >= 0) ? 1 : 0;
    src_out[(size_t)g * cE + e] = ns < 0 ? 0 : ns;
    dst_out[(size_t)g * cE + e] = nd < 0 ? 0 : nd;
    msk_out[(size_t)g * cE + e] = m;
  }
}

// ======================================================================
// Layers 2/3: GraphConv(128->128) + ReLU + TopK + readout (padded to 64 rows)
// Input tile staged to LDS by the Tensor Data Mover when available.
// ======================================================================
__global__ __launch_bounds__(256)
void k_layerN(const bf16* __restrict__ xin, int n_in, int k_out,
              const bf16* __restrict__ Wr, const float* __restrict__ bias,
              const bf16* __restrict__ Wo, const float* __restrict__ p,
              const int* __restrict__ src_in, const int* __restrict__ dst_in,
              const int* __restrict__ msk_in,
              int* __restrict__ src_out, int* __restrict__ dst_out, int* __restrict__ msk_out,
              bf16* __restrict__ xout, float* __restrict__ h) {
  __shared__ __align__(16) unsigned char smem[99360];
  bf16*  sXbf   = (bf16*)smem;                       // [64][128]
  bf16*  sAggBf = (bf16*)(smem + 16384);             // [64][128]
  float* sAgg   = (float*)(smem + 32768);            // [64][128] f32
  float* sY     = (float*)(smem + 32768);            // aliases sAgg
  float* sXk    = (float*)(smem + 65536);            // [64][128]
  float* sS     = (float*)(smem + 98304);
  float* sVal   = (float*)(smem + 98304 + 256);
  int*   sIdx   = (int*)  (smem + 98304 + 512);
  int*   sInv   = (int*)  (smem + 98304 + 768);
  float* sPn    = (float*)(smem + 98304 + 1024);

  const int g = blockIdx.x;
  const int tid = threadIdx.x;
  const int lane = tid & 31, wv = tid >> 5;

#if HAVE_TDM
  // Stage xin tile (64x128 bf16 = 16KB contiguous) via TDM DMA; wave 0 issues
  // one descriptor (TDM ignores EXEC; one D# per workgroup avoids dup traffic).
  for (int idx = tid; idx < 64 * 128; idx += 256) sAgg[idx] = 0.f;
  if (tid < 32) {
    unsigned lds_base = (unsigned)(uintptr_t)(void*)&smem[0];   // aperture low bits = LDS offset
    unsigned long long ga = (unsigned long long)(uintptr_t)(xin + (size_t)g * 64 * 128);
    u32x4 g0;
    g0[0] = 1u;                                       // count=1, user mode
    g0[1] = lds_base;                                 // lds_addr (bytes)
    g0[2] = (unsigned)(ga & 0xFFFFFFFFu);             // global_addr[31:0]
    g0[3] = (unsigned)((ga >> 32) & 0x01FFFFFFu) | (2u << 30);  // addr[56:32] | type=2
    i32x8 g1;
    g1[0] = (1 << 16);        // workgroup_mask=0 | data_size=1 (2 bytes)
    g1[1] = (8192 << 16);     // atomic_barrier=0 | tensor_dim0 lo16 = 8192
    g1[2] = (1 << 16);        // tensor_dim0 hi16=0 | tensor_dim1 lo16 = 1
    g1[3] = (8192 << 16);     // tensor_dim1 hi16=0 | tile_dim0 = 8192
    g1[4] = 1;                // tile_dim1=1, tile_dim2=0
    g1[5] = 8192;             // tensor_dim0_stride lo32
    g1[6] = 0;                // stride hi16 | tensor_dim1_stride lo16
    g1[7] = 0;                // tensor_dim1_stride hi32
    i32x4 z4 = {0, 0, 0, 0};
#if __clang_major__ >= 23
    i32x8 z8 = {0, 0, 0, 0, 0, 0, 0, 0};
    __builtin_amdgcn_tensor_load_to_lds(g0, g1, z4, z4, z8, 0);
#else
    __builtin_amdgcn_tensor_load_to_lds(g0, g1, z4, z4, 0);
#endif
    __builtin_amdgcn_s_wait_tensorcnt(0);
  }
  __syncthreads();
#else
  for (int idx = tid; idx < 64 * 128; idx += 256) {
    sXbf[idx] = xin[(size_t)g * 64 * 128 + idx];
    sAgg[idx] = 0.f;
  }
  __syncthreads();
#endif

  for (int idx = tid; idx < cE * 128; idx += 256) {
    int e = idx >> 7, c = idx & 127;
    if (msk_in[(size_t)g * cE + e]) {
      int s0 = src_in[(size_t)g * cE + e];
      int d0 = dst_in[(size_t)g * cE + e];
      atomicAdd(&sAgg[d0 * 128 + c], bf2f(sXbf[s0 * 128 + c]));
    }
  }
  __syncthreads();
  for (int i = tid; i < 64 * 128; i += 256) sAggBf[i] = f2bf(sAgg[i]);
  __syncthreads();

  for (int t = wv; t < 32; t += 8) {
    int mt = t >> 3, nt = t & 7;
    v8f acc = {};
#pragma unroll
    for (int kk = 0; kk < 128; kk += 32) {
      acc = wmma_bf16(load_a_frag(sAggBf, 128, mt * 16, kk, lane),
                      load_b_frag(Wr, 128, kk, nt * 16, lane), acc);
      acc = wmma_bf16(load_a_frag(sXbf, 128, mt * 16, kk, lane),
                      load_b_frag(Wo, 128, kk, nt * 16, lane), acc);
    }
    int n = nt * 16 + (lane & 15);
    int mb = mt * 16 + (lane >> 4) * 8;
    float bv = bias[n];
#pragma unroll
    for (int r = 0; r < 8; ++r) {
      float v = acc[r] + bv;
      sY[(mb + r) * 128 + n] = v > 0.f ? v : 0.f;
    }
  }
  __syncthreads();

  if (tid == 0) {
    float s = 0.f;
    for (int c = 0; c < 128; ++c) s += p[c] * p[c];
    *sPn = sqrtf(s);
  }
  __syncthreads();
  if (tid < 64) {
    float s = 0.f;
    if (tid < n_in)
      for (int c = 0; c < 128; ++c) s += sY[tid * 128 + c] * p[c];
    sS[tid] = (tid < n_in) ? s / *sPn : -1e30f;
  }
  __syncthreads();
  if (tid == 0) {
    for (int j = 0; j < 64; ++j) sInv[j] = -1;
    for (int i = 0; i < k_out; ++i) {
      float best = -1e30f; int bj = 0;
      for (int j = 0; j < n_in; ++j) if (sS[j] > best) { best = sS[j]; bj = j; }
      sIdx[i] = bj; sVal[i] = best; sInv[bj] = i; sS[bj] = -1e30f;
    }
  }
  __syncthreads();

  for (int idx = tid; idx < 64 * 128; idx += 256) {
    int i = idx >> 7, c = idx & 127;
    float v = 0.f;
    if (i < k_out) v = sY[sIdx[i] * 128 + c] * tanhf(sVal[i]);
    sXk[idx] = v;
    xout[(size_t)g * 64 * 128 + idx] = f2bf(v);
  }
  __syncthreads();

  if (tid < 128) {
    int c = tid;
    float mx = -1e30f, sm = 0.f;
    for (int i = 0; i < k_out; ++i) { float v = sXk[i * 128 + c]; mx = fmaxf(mx, v); sm += v; }
    h[(size_t)g * 256 + c]       += mx;
    h[(size_t)g * 256 + 128 + c] += sm / (float)k_out;
  }

  for (int e = tid; e < cE; e += 256) {
    int s0 = src_in[(size_t)g * cE + e];
    int d0 = dst_in[(size_t)g * cE + e];
    int mi = msk_in[(size_t)g * cE + e];
    int ns = sInv[s0], nd = sInv[d0];
    int m = (mi && ns >= 0 && nd >= 0) ? 1 : 0;
    src_out[(size_t)g * cE + e] = ns < 0 ? 0 : ns;
    dst_out[(size_t)g * cE + e] = nd < 0 ? 0 : nd;
    msk_out[(size_t)g * cE + e] = m;
  }
}

// ======================================================================
// MLP head: hf = relu(relu(h@L1+b1)@L2+b2), 16 graph-rows per block, WMMA
// ======================================================================
__global__ __launch_bounds__(256)
void k_mlp(const float* __restrict__ h,
           const bf16* __restrict__ L1, const float* __restrict__ L1b,
           const bf16* __restrict__ L2, const float* __restrict__ L2b,
           float* __restrict__ hf) {
  __shared__ bf16 sA [16 * 256];
  __shared__ bf16 sA2[16 * 256];
  const int g0 = blockIdx.x * 16;
  const int tid = threadIdx.x, lane = tid & 31, wv = tid >> 5;

  for (int idx = tid; idx < 16 * 256; idx += 256)
    sA[idx] = f2bf(h[(size_t)g0 * 256 + idx]);
  __syncthreads();

  for (int nt = wv; nt < 16; nt += 8) {         // lin1: 256 out
    v8f acc = {};
#pragma unroll
    for (int kk = 0; kk < 256; kk += 32)
      acc = wmma_bf16(load_a_frag(sA, 256, 0, kk, lane),
                      load_b_frag(L1, 256, kk, nt * 16, lane), acc);
    int n = nt * 16 + (lane & 15);
    int mb = (lane >> 4) * 8;
    float bv = L1b[n];
#pragma unroll
    for (int r = 0; r < 8; ++r) {
      float v = acc[r] + bv; v = v > 0.f ? v : 0.f;
      sA2[(mb + r) * 256 + n] = f2bf(v);
    }
  }
  __syncthreads();

  for (int nt = wv; nt < 8; nt += 8) {          // lin2: 128 out
    v8f acc = {};
#pragma unroll
    for (int kk = 0; kk < 256; kk += 32)
      acc = wmma_bf16(load_a_frag(sA2, 256, 0, kk, lane),
                      load_b_frag(L2, 128, kk, nt * 16, lane), acc);
    int n = nt * 16 + (lane & 15);
    int mb = (lane >> 4) * 8;
    float bv = L2b[n];
#pragma unroll
    for (int r = 0; r < 8; ++r) {
      float v = acc[r] + bv;
      hf[(size_t)(g0 + mb + r) * 128 + n] = v > 0.f ? v : 0.f;
    }
  }
}

// ======================================================================
// Final scoring: sigmoid(dot(emb_item[item_id[g,n]], hf[g]))
// ======================================================================
__global__ __launch_bounds__(256)
void k_score(const int* __restrict__ item_id, const float* __restrict__ emb_item,
             const float* __restrict__ hf, float* __restrict__ out) {
  int t = blockIdx.x * blockDim.x + threadIdx.x;
  if (t >= cG * cN) return;
  int g = t >> 6;
  const float* e  = emb_item + (size_t)item_id[t] * cD;
  const float* hh = hf + (size_t)g * cD;
  __builtin_prefetch(e, 0, 1);
  float s = 0.f;
#pragma unroll 8
  for (int c = 0; c < 128; ++c) s += e[c] * hh[c];
  out[t] = 1.f / (1.f + expf(-s));
}

// ======================================================================
extern "C" void kernel_launch(void* const* d_in, const int* in_sizes, int n_in_,
                              void* d_out, int out_size, void* d_ws, size_t ws_size,
                              hipStream_t stream) {
  (void)in_sizes; (void)n_in_; (void)out_size; (void)ws_size;
  const int*   item_id    = (const int*)d_in[0];
  const int*   category   = (const int*)d_in[1];
  const int*   edge_local = (const int*)d_in[2];
  const float* emb_item   = (const float*)d_in[3];
  const float* emb_cat    = (const float*)d_in[4];
  const float* W1r = (const float*)d_in[5];  const float* b1 = (const float*)d_in[6];
  const float* W1o = (const float*)d_in[7];
  const float* W2r = (const float*)d_in[8];  const float* b2 = (const float*)d_in[9];
  const float* W2o = (const float*)d_in[10];
  const float* W3r = (const float*)d_in[11]; const float* b3 = (const float*)d_in[12];
  const float* W3o = (const float*)d_in[13];
  const float* p1  = (const float*)d_in[14]; const float* p2 = (const float*)d_in[15];
  const float* p3  = (const float*)d_in[16];
  const float* L1W = (const float*)d_in[17]; const float* L1b = (const float*)d_in[18];
  const float* L2W = (const float*)d_in[19]; const float* L2b = (const float*)d_in[20];
  float* out = (float*)d_out;

  // workspace carving (256B aligned)
  size_t off = 0;
  char* wsb = (char*)d_ws;
  auto carve = [&](size_t bytes) -> void* {
    void* pp = wsb + off;
    off = (off + bytes + 255) & ~(size_t)255;
    return pp;
  };
  bf16* W1r_bf = (bf16*)carve((size_t)256 * 128 * 2);
  bf16* W1o_bf = (bf16*)carve((size_t)256 * 128 * 2);
  bf16* W2r_bf = (bf16*)carve((size_t)128 * 128 * 2);
  bf16* W2o_bf = (bf16*)carve((size_t)128 * 128 * 2);
  bf16* W3r_bf = (bf16*)carve((size_t)128 * 128 * 2);
  bf16* W3o_bf = (bf16*)carve((size_t)128 * 128 * 2);
  bf16* L1_bf  = (bf16*)carve((size_t)256 * 256 * 2);
  bf16* L2_bf  = (bf16*)carve((size_t)256 * 128 * 2);
  bf16* xA = (bf16*)carve((size_t)cG * 64 * 128 * 2);
  bf16* xB = (bf16*)carve((size_t)cG * 64 * 128 * 2);
  int* srcA = (int*)carve((size_t)cG * cE * 4);
  int* dstA = (int*)carve((size_t)cG * cE * 4);
  int* mskA = (int*)carve((size_t)cG * cE * 4);
  int* srcB = (int*)carve((size_t)cG * cE * 4);
  int* dstB = (int*)carve((size_t)cG * cE * 4);
  int* mskB = (int*)carve((size_t)cG * cE * 4);
  float* h  = (float*)carve((size_t)cG * 256 * 4);
  float* hf = (float*)carve((size_t)cG * 128 * 4);

  auto cvt = [&](const float* s, bf16* d, int n) {
    k_cvt<<<(n + 255) / 256, 256, 0, stream>>>(s, d, n);
  };
  cvt(W1r, W1r_bf, 256 * 128); cvt(W1o, W1o_bf, 256 * 128);
  cvt(W2r, W2r_bf, 128 * 128); cvt(W2o, W2o_bf, 128 * 128);
  cvt(W3r, W3r_bf, 128 * 128); cvt(W3o, W3o_bf, 128 * 128);
  cvt(L1W, L1_bf, 256 * 256);  cvt(L2W, L2_bf, 256 * 128);

  k_zero<<<(cG * 256 + 255) / 256, 256, 0, stream>>>(h, cG * 256);

  k_layer1<<<cG, 256, 0, stream>>>(item_id, category, edge_local, emb_item, emb_cat,
                                   W1r_bf, b1, W1o_bf, p1,
                                   srcA, dstA, mskA, xA, h);
  k_layerN<<<cG, 256, 0, stream>>>(xA, KP1, KP2, W2r_bf, b2, W2o_bf, p2,
                                   srcA, dstA, mskA, srcB, dstB, mskB, xB, h);
  k_layerN<<<cG, 256, 0, stream>>>(xB, KP2, KP3, W3r_bf, b3, W3o_bf, p3,
                                   srcB, dstB, mskB, srcA, dstA, mskA, xA, h);

  k_mlp<<<cG / 16, 256, 0, stream>>>(h, L1_bf, L1b, L2_bf, L2b, hf);
  k_score<<<(cG * cN + 255) / 256, 256, 0, stream>>>(item_id, emb_item, hf, out);
}
